// Multi_head_sequence_embedding_26371099197827
// MI455X (gfx1250) — matrix operand order repaired and checked by reading, not compiled, not run on hardware
//
#include <hip/hip_runtime.h>

#define DD 256
#define NN 1024
#define BB 32
#define MM (BB * NN)   // 32768 total rows

#define KP 40    // padded LDS row stride (halves) for K tile: 80B -> 2-way max conflict
#define VP 264   // padded LDS row stride (halves) for V tile: 528B -> 2-way max conflict

typedef _Float16 v16h __attribute__((ext_vector_type(16)));
typedef _Float16 v8h  __attribute__((ext_vector_type(8)));
typedef float    v8f  __attribute__((ext_vector_type(8)));
typedef int      v4i  __attribute__((ext_vector_type(4)));

#ifndef __has_builtin
#define __has_builtin(x) 0
#endif

#if __has_builtin(__builtin_amdgcn_global_load_async_to_lds_b128)
#define HAVE_ASYNC_LDS 1
#else
#define HAVE_ASYNC_LDS 0
#endif

// AS1: global pointers keep their 64-bit pattern in generic space.
// AS3: generic LDS pointer = {aperture, offset[31:0]}; low 32 bits are the DS offset.
#define AS1V4(p) ((__attribute__((address_space(1))) v4i*)(unsigned long long)(p))
#define AS3V4(p) ((__attribute__((address_space(3))) v4i*)(unsigned int)(unsigned long long)(p))

static __device__ __forceinline__ v16h cat8(v8h lo, v8h hi) {
    return __builtin_shufflevector(lo, hi, 0,1,2,3,4,5,6,7,8,9,10,11,12,13,14,15);
}

static __device__ __forceinline__ v8f wmma_f16(v16h a, v16h b, v8f c) {
    return __builtin_amdgcn_wmma_f32_16x16x32_f16(false, a, false, b, (short)0, c, false, false);
}

// 16-byte global -> LDS copy: async path (ASYNCcnt) if available, else load + ds_store.
static __device__ __forceinline__ void copy16(const _Float16* g, _Float16* l) {
#if HAVE_ASYNC_LDS
    __builtin_amdgcn_global_load_async_to_lds_b128(AS1V4(g), AS3V4(l), 0, 0);
#else
    *(v8h*)l = *(const v8h*)g;
#endif
}

static __device__ __forceinline__ void wait_async() {
#if HAVE_ASYNC_LDS
#if __has_builtin(__builtin_amdgcn_s_wait_asynccnt)
    __builtin_amdgcn_s_wait_asynccnt(0);
#else
    asm volatile("s_wait_asynccnt 0x0" ::: "memory");
#endif
#endif
}

// ---------------------------------------------------------------- prep kernels
__global__ void msq_cvt_x(const float* __restrict__ x, _Float16* __restrict__ y, int n) {
    int i = blockIdx.x * blockDim.x + threadIdx.x;
    if (i < n) y[i] = (_Float16)x[i];
}

// Wt[k][e] = W[e][k] as f16, for the three weight matrices.
__global__ void msq_cvt_w(const float* __restrict__ Wq, const float* __restrict__ Wk,
                          const float* __restrict__ Wv, _Float16* __restrict__ Wtq,
                          _Float16* __restrict__ Wtk, _Float16* __restrict__ Wtv) {
    int i = blockIdx.x * blockDim.x + threadIdx.x;   // 3 * 65536
    int mat = i >> 16;
    int r = i & 65535;
    int e = r & (DD - 1);
    int k = r >> 8;
    const float* W   = (mat == 0) ? Wq : (mat == 1) ? Wk : Wv;
    _Float16*    Wt  = (mat == 0) ? Wtq : (mat == 1) ? Wtk : Wtv;
    Wt[k * DD + e] = (_Float16)W[e * DD + k];
}

// ---------------------------------------------------------------- QKV projection
// One wave per 16-row tile of X; computes Q, K(transposed), V, f16 outputs.
__global__ void msq_qkv(const _Float16* __restrict__ X,
                        const _Float16* __restrict__ Wtq, const _Float16* __restrict__ Wtk,
                        const _Float16* __restrict__ Wtv,
                        const float* __restrict__ bq, const float* __restrict__ bk,
                        const float* __restrict__ bv,
                        _Float16* __restrict__ Q, _Float16* __restrict__ Kt,
                        _Float16* __restrict__ V) {
    const int lane = threadIdx.x & 31;
    const int wave = threadIdx.x >> 5;
    const int tile = blockIdx.x * (blockDim.x >> 5) + wave;   // 0..2047
    const int mb   = tile * 16;
    const int half = lane >> 4;
    const int mr   = lane & 15;   // A-frag row / C-frag column index

    // resident A fragments of the 16x256 X tile
    v16h xa[8];
    const _Float16* xrow = X + (size_t)(mb + mr) * DD;
    #pragma unroll
    for (int s = 0; s < 8; ++s) {
        v8h lo = *(const v8h*)(xrow + 32 * s + half * 8);
        v8h hi = *(const v8h*)(xrow + 32 * s + 16 + half * 8);
        xa[s] = cat8(lo, hi);
    }

    const int bidx = mb >> 10;        // batch index
    const int nb   = mb & (NN - 1);   // sequence base within batch

    #pragma unroll 1
    for (int mat = 0; mat < 3; ++mat) {
        const _Float16* Wt  = (mat == 0) ? Wtq : (mat == 1) ? Wtk : Wtv;
        const float*    bia = (mat == 0) ? bq : (mat == 1) ? bk : bv;
        #pragma unroll 1
        for (int t = 0; t < 16; ++t) {
            v8f acc = {};
            #pragma unroll
            for (int s = 0; s < 8; ++s) {
                v16h wb;
                #pragma unroll
                for (int v = 0; v < 16; ++v)
                    wb[v] = Wt[(size_t)(32 * s + (v & 7) + 8 * half + 16 * (v >> 3)) * DD + t * 16 + mr];
                acc = wmma_f16(xa[s], wb, acc);
            }
            float bval = bia[t * 16 + mr];   // column = lane&15
            if (mat == 1) {
                // K stored transposed: Kt[b][e][n]; 8 consecutive n per lane -> one 16B store
                v8h kv;
                #pragma unroll
                for (int v = 0; v < 8; ++v) kv[v] = (_Float16)(acc[v] + bval);
                _Float16* kp = Kt + (size_t)bidx * DD * NN + (size_t)(t * 16 + mr) * NN
                                  + nb + 8 * half;
                *(v8h*)kp = kv;
            } else {
                _Float16* O = (mat == 0) ? Q : V;
                #pragma unroll
                for (int v = 0; v < 8; ++v)
                    O[(size_t)(mb + v + 8 * half) * DD + t * 16 + mr] =
                        (_Float16)(acc[v] + bval);
            }
        }
    }
}

// ---------------------------------------------------------------- fused attention
// Block = 8 waves, all on the SAME batch (64 row-tiles/batch, 8 tiles/block).
// Per 32-key-column step the block stages the K tile (256x32) and V tile (32x256)
// into LDS once (async-to-LDS when available); every wave consumes via ds_load.
__global__ void msq_attn(const _Float16* __restrict__ Q, const _Float16* __restrict__ Kt,
                         const _Float16* __restrict__ V, float* __restrict__ out) {
    __shared__ __align__(64) _Float16 kbuf[256 * KP];    // 20.0 KB
    __shared__ __align__(64) _Float16 vbuf[32 * VP];     // 16.5 KB
    __shared__ __align__(64) _Float16 pbuf[8][16 * 32];  //  8.0 KB

    const int tid  = threadIdx.x;
    const int lane = tid & 31;
    const int w    = tid >> 5;
    const int tile = blockIdx.x * 8 + w;                 // 0..2047
    const int b    = tile >> 6;                          // batch (8 | 64 -> no straddle)
    const int ib   = (tile & 63) * 16;                   // query row base in batch
    const int half = lane >> 4;
    const int mr   = lane & 15;

    const _Float16* Ktb = Kt + (size_t)b * DD * NN;
    const _Float16* Vb  = V + (size_t)b * NN * DD;

    // resident Q A-fragments (16 rows x 256 d)
    v16h qa[8];
    const _Float16* qrow = Q + (size_t)(b * NN + ib + mr) * DD;
    #pragma unroll
    for (int s = 0; s < 8; ++s) {
        v8h lo = *(const v8h*)(qrow + 32 * s + half * 8);
        v8h hi = *(const v8h*)(qrow + 32 * s + 16 + half * 8);
        qa[s] = cat8(lo, hi);
    }

    v8f o[16];
    #pragma unroll
    for (int c = 0; c < 16; ++c) o[c] = (v8f){};
    float mrun[8], lrun[8];
    #pragma unroll
    for (int v = 0; v < 8; ++v) { mrun[v] = -1e30f; lrun[v] = 0.0f; }

    #pragma unroll 1
    for (int j = 0; j < NN; j += 32) {
        __syncthreads();   // previous tile fully consumed by all waves

        // ---- stage K tile: 256 d-rows x 32 key cols (4x16B per thread)
        #pragma unroll
        for (int r = 0; r < 4; ++r) {
            int d = (tid >> 2) + 64 * r, ch = tid & 3;
            copy16(Ktb + (size_t)d * NN + j + ch * 8, kbuf + d * KP + ch * 8);
        }
        // ---- stage V tile: 32 key rows x 256 d (4x16B per thread)
        #pragma unroll
        for (int r = 0; r < 4; ++r) {
            int u = tid + 256 * r, row = u >> 5, ch = u & 31;
            copy16(Vb + (size_t)(j + row) * DD + ch * 8, vbuf + row * VP + ch * 8);
        }
        wait_async();
        if (j + 32 < NN) {  // prefetch next tiles toward L2/L0
            __builtin_prefetch(Ktb + (size_t)tid * NN + j + 32, 0, 1);
            __builtin_prefetch(Vb + (size_t)(j + 32 + (tid >> 3)) * DD + (tid & 7) * 32, 0, 1);
        }
        __syncthreads();

        // ---- S = Q * K^T for 32 key columns, fp32 accumulate, B-frags from LDS
        v8f s0 = (v8f){}, s1 = (v8f){};
        #pragma unroll
        for (int s = 0; s < 8; ++s) {
            v16h kb0, kb1;
            #pragma unroll
            for (int v = 0; v < 16; ++v) {
                const _Float16* kp = kbuf + (32 * s + (v & 7) + 8 * half + 16 * (v >> 3)) * KP;
                kb0[v] = kp[mr];
                kb1[v] = kp[16 + mr];
            }
            s0 = wmma_f16(qa[s], kb0, s0);
            s1 = wmma_f16(qa[s], kb1, s1);
        }

        // ---- online softmax: row reductions across 16 lanes of each half-wave
        float mnew[8], p0[8], p1[8], scl[8];
        #pragma unroll
        for (int v = 0; v < 8; ++v) {
            float t = fmaxf(s0[v], s1[v]);
            t = fmaxf(t, __shfl_xor(t, 1, 16));
            t = fmaxf(t, __shfl_xor(t, 2, 16));
            t = fmaxf(t, __shfl_xor(t, 4, 16));
            t = fmaxf(t, __shfl_xor(t, 8, 16));
            mnew[v] = fmaxf(mrun[v], t);
            scl[v]  = __expf(mrun[v] - mnew[v]);
            p0[v]   = __expf(s0[v] - mnew[v]);
            p1[v]   = __expf(s1[v] - mnew[v]);
            float r = p0[v] + p1[v];
            r += __shfl_xor(r, 1, 16);
            r += __shfl_xor(r, 2, 16);
            r += __shfl_xor(r, 4, 16);
            r += __shfl_xor(r, 8, 16);
            lrun[v] = lrun[v] * scl[v] + r;
            mrun[v] = mnew[v];
        }
        #pragma unroll
        for (int c = 0; c < 16; ++c)
            #pragma unroll
            for (int v = 0; v < 8; ++v) o[c][v] *= scl[v];

        // ---- repack P from C-layout (lane=col) to A-layout (lane=row) via LDS
        _Float16* pb = &pbuf[w][0];
        #pragma unroll
        for (int v = 0; v < 8; ++v) {
            pb[(v + 8 * half) * 32 + mr]      = (_Float16)p0[v];
            pb[(v + 8 * half) * 32 + 16 + mr] = (_Float16)p1[v];
        }
        __syncthreads();   // uniform trip counts across all waves
        v8h plo = *(const v8h*)(pb + mr * 32 + half * 8);
        v8h phi = *(const v8h*)(pb + mr * 32 + 16 + half * 8);
        v16h pa = cat8(plo, phi);

        // ---- O += P * V, V B-frags from LDS
        #pragma unroll
        for (int c = 0; c < 16; ++c) {
            v16h vb;
            #pragma unroll
            for (int v = 0; v < 16; ++v)
                vb[v] = vbuf[((v & 7) + 8 * half + 16 * (v >> 3)) * VP + c * 16 + mr];
            o[c] = wmma_f16(pa, vb, o[c]);
        }
    }

    // ---- normalize and store fp32 output
    float rinv[8];
    #pragma unroll
    for (int v = 0; v < 8; ++v) rinv[v] = 1.0f / lrun[v];
    float* orow = out + (size_t)(b * NN + ib) * DD;
    #pragma unroll
    for (int c = 0; c < 16; ++c)
        #pragma unroll
        for (int v = 0; v < 8; ++v)
            orow[(size_t)(v + 8 * half) * DD + c * 16 + mr] = o[c][v] * rinv[v];
}

// ---------------------------------------------------------------- launcher
extern "C" void kernel_launch(void* const* d_in, const int* in_sizes, int n_in,
                              void* d_out, int out_size, void* d_ws, size_t ws_size,
                              hipStream_t stream) {
    const float* traj = (const float*)d_in[0];
    const float* Wq   = (const float*)d_in[1];
    const float* bq   = (const float*)d_in[2];
    const float* Wk   = (const float*)d_in[3];
    const float* bk   = (const float*)d_in[4];
    const float* Wv   = (const float*)d_in[5];
    const float* bv   = (const float*)d_in[6];
    float* out = (float*)d_out;

    char* ws = (char*)d_ws;
    const size_t SZ_MD = (size_t)MM * DD * sizeof(_Float16);   // 16.78 MB each
    _Float16* X16 = (_Float16*)(ws);
    _Float16* Q16 = (_Float16*)(ws + SZ_MD);
    _Float16* Kt  = (_Float16*)(ws + 2 * SZ_MD);
    _Float16* V16 = (_Float16*)(ws + 3 * SZ_MD);
    _Float16* Wtq = (_Float16*)(ws + 4 * SZ_MD);
    _Float16* Wtk = (_Float16*)(ws + 4 * SZ_MD + (size_t)DD * DD * 2);
    _Float16* Wtv = (_Float16*)(ws + 4 * SZ_MD + (size_t)DD * DD * 4);

    // 1. convert activations
    msq_cvt_x<<<(MM * DD + 255) / 256, 256, 0, stream>>>(traj, X16, MM * DD);
    // 2. transpose + convert weights
    msq_cvt_w<<<(3 * DD * DD + 255) / 256, 256, 0, stream>>>(Wq, Wk, Wv, Wtq, Wtk, Wtv);
    // 3. QKV projection: 2048 tiles, 8 waves per block
    msq_qkv<<<MM / 16 / 8, 256, 0, stream>>>(X16, Wtq, Wtk, Wtv, bq, bk, bv, Q16, Kt, V16);
    // 4. fused flash attention: 2048 tiles, 8 waves per block, LDS-shared K/V tiles
    msq_attn<<<MM / 16 / 8, 256, 0, stream>>>(Q16, Kt, V16, out);
}
